// DeformableTransformerEncoderLayer_83373905150543
// MI455X (gfx1250) — matrix operand order repaired and checked by reading, not compiled, not run on hardware
//
#include <hip/hip_runtime.h>
#include <hip/hip_bf16.h>

// ---------------- static problem geometry (from reference) ----------------
#define DMODEL 256
#define NHEAD  8
#define NLVL   4
#define NPTS   4
#define DHEAD  32
#define DFFN   1024
#define QLEN   5440
#define BATCH  4
#define MROWS  (BATCH*QLEN)   // 21760 = 64*340
#define EPSLN  1e-5f

typedef __bf16 bf16_t;
typedef __bf16 v16bf __attribute__((ext_vector_type(16)));
typedef __bf16 v8bf  __attribute__((ext_vector_type(8)));
typedef float  v8f   __attribute__((ext_vector_type(8)));

// ---------------- prep: bf16 copies of src and query = src + pos ----------
__global__ __launch_bounds__(256) void prep_kernel(const float* __restrict__ src,
                                                   const float* __restrict__ pos,
                                                   bf16_t* __restrict__ src_b,
                                                   bf16_t* __restrict__ q_b, int n) {
  int i = blockIdx.x * blockDim.x + threadIdx.x;
  if (i < n) {
    float s = src[i];
    src_b[i] = (bf16_t)s;
    q_b[i]   = (bf16_t)(s + pos[i]);
  }
}

// ---------------- weight convert + transpose: W[K,N] f32 -> Wt[N,K] bf16 --
__global__ __launch_bounds__(256) void wtrans_kernel(const float* __restrict__ W,
                                                     bf16_t* __restrict__ Wt,
                                                     int K, int N) {
  int i = blockIdx.x * blockDim.x + threadIdx.x;
  if (i < K * N) {
    int k = i / N, n = i % N;
    Wt[(size_t)n * K + k] = (bf16_t)W[i];
  }
}

// ---------------- bf16 WMMA GEMM: out[M,N] = A[M,K] * Wt[N,K]^T + bias ----
// Each wave computes a 32x32 output patch as a 2x2 grid of 16x16 WMMA tiles:
// 4 fragment loads feed 4 v_wmma_f32_16x16x32_bf16 per K-step (A and B each
// reused twice from registers). Block = 8 waves = 64 rows x 128 cols.
// Epilogue: optional residual add (f32), optional ReLU, f32 and/or bf16 out.
__global__ __launch_bounds__(256) void gemm_wmma_kernel(
    const bf16_t* __restrict__ A, const bf16_t* __restrict__ Wt,
    const float* __restrict__ bias, int M, int N, int K,
    float* outF, bf16_t* outB,
    const float* __restrict__ residual, int relu)
{
  int lane = threadIdx.x & 31;
  int wid  = threadIdx.x >> 5;
  int hl   = lane >> 4;          // half-wave select
  int r    = lane & 15;
  int wm   = wid >> 2;           // 0..1 : M group within block
  int wn   = wid & 3;            // 0..3 : N group within block
  int m_base = blockIdx.y * 64 + wm * 32;
  int n_base = blockIdx.x * 128 + wn * 32;

  const bf16_t* arow0 = A  + (size_t)(m_base + r) * K;       // rows m..m+15
  const bf16_t* arow1 = A  + (size_t)(m_base + 16 + r) * K;  // rows m+16..m+31
  const bf16_t* brow0 = Wt + (size_t)(n_base + r) * K;       // cols n..n+15
  const bf16_t* brow1 = Wt + (size_t)(n_base + 16 + r) * K;  // cols n+16..n+31

  v8f acc00 = {}, acc01 = {}, acc10 = {}, acc11 = {};
  for (int k0 = 0; k0 < K; k0 += 32) {
    union { v16bf v; v8bf h[2]; } a0, a1, b0, b1;
    // A 16x32 layout: elem e <-> K = k0 + 16*hl + e
    a0.h[0] = *(const v8bf*)(arow0 + k0 + 16 * hl);
    a0.h[1] = *(const v8bf*)(arow0 + k0 + 16 * hl + 8);
    a1.h[0] = *(const v8bf*)(arow1 + k0 + 16 * hl);
    a1.h[1] = *(const v8bf*)(arow1 + k0 + 16 * hl + 8);
    // B 32x16 layout: elem e <-> K = k0 + 16*hl + e (contiguous in [N,K])
    b0.h[0] = *(const v8bf*)(brow0 + k0 + 16 * hl);
    b0.h[1] = *(const v8bf*)(brow0 + k0 + 16 * hl + 8);
    b1.h[0] = *(const v8bf*)(brow1 + k0 + 16 * hl);
    b1.h[1] = *(const v8bf*)(brow1 + k0 + 16 * hl + 8);
    acc00 = __builtin_amdgcn_wmma_f32_16x16x32_bf16(false, a0.v, false, b0.v,
                                                    (short)0, acc00, false, false);
    acc01 = __builtin_amdgcn_wmma_f32_16x16x32_bf16(false, a0.v, false, b1.v,
                                                    (short)0, acc01, false, false);
    acc10 = __builtin_amdgcn_wmma_f32_16x16x32_bf16(false, a1.v, false, b0.v,
                                                    (short)0, acc10, false, false);
    acc11 = __builtin_amdgcn_wmma_f32_16x16x32_bf16(false, a1.v, false, b1.v,
                                                    (short)0, acc11, false, false);
  }

  auto store_tile = [&](v8f acc, int m0, int col) {
    float bsc = bias ? bias[col] : 0.0f;
    #pragma unroll
    for (int i = 0; i < 8; ++i) {
      int row = m0 + i + 8 * hl;   // C/D layout: VGPR i -> M = i + 8*half
      float v = acc[i] + bsc;
      if (residual) v += residual[(size_t)row * N + col];
      if (relu) v = v > 0.0f ? v : 0.0f;
      if (outF) outF[(size_t)row * N + col] = v;
      if (outB) outB[(size_t)row * N + col] = (bf16_t)v;
    }
  };
  store_tile(acc00, m_base,      n_base + r);
  store_tile(acc01, m_base,      n_base + 16 + r);
  store_tile(acc10, m_base + 16, n_base + r);
  store_tile(acc11, m_base + 16, n_base + 16 + r);
}

// ---------------- deformable sampling: one wave per (b, q, head) ----------
// lane = channel (DHEAD == 32 == wave32). Softmax over NLVL*NPTS=16 weights
// is computed redundantly per lane from the logits GEMM output.
__global__ __launch_bounds__(256) void msdeform_sample_kernel(
    const bf16_t* __restrict__ value,   // [MROWS, 256] (head-major inner)
    const float*  __restrict__ off,     // [MROWS, 256]
    const float*  __restrict__ logits,  // [MROWS, 128]
    bf16_t* __restrict__ attn_out)      // [MROWS, 256]
{
  const int LHW[4] = {64, 32, 16, 8};
  const int LS[4]  = {0, 4096, 5120, 5376};

  int lane = threadIdx.x & 31;
  int gw = blockIdx.x * 8 + (threadIdx.x >> 5);   // 0 .. 174079
  int h = gw & 7;
  int q = (gw >> 3) % QLEN;
  int b = gw / (8 * QLEN);
  int row = b * QLEN + q;

  // reference point from the query's own level (all levels are square)
  int lq, pq;
  if      (q < 4096) { lq = 0; pq = q; }
  else if (q < 5120) { lq = 1; pq = q - 4096; }
  else if (q < 5376) { lq = 2; pq = q - 5120; }
  else               { lq = 3; pq = q - 5376; }
  int Wq = LHW[lq];
  float refx = ((float)(pq % Wq) + 0.5f) / (float)Wq;
  float refy = ((float)(pq / Wq) + 0.5f) / (float)Wq;

  // softmax over 16 attention logits for this head
  const float* lg = logits + (size_t)row * 128 + h * 16;
  float w16[16], mx = -1e30f;
  #pragma unroll
  for (int j = 0; j < 16; ++j) { w16[j] = lg[j]; mx = w16[j] > mx ? w16[j] : mx; }
  float ssum = 0.0f;
  #pragma unroll
  for (int j = 0; j < 16; ++j) { w16[j] = __expf(w16[j] - mx); ssum += w16[j]; }
  float sinv = 1.0f / ssum;

  const float* orow = off + (size_t)row * 256;
  float acc = 0.0f;
  #pragma unroll
  for (int l = 0; l < 4; ++l) {
    const int HW = LHW[l], S = LS[l];
    #pragma unroll
    for (int p = 0; p < 4; ++p) {
      int oidx = (((h * 4 + l) * 4 + p) * 2);
      float x = refx * (float)HW + orow[oidx]     - 0.5f;
      float y = refy * (float)HW + orow[oidx + 1] - 0.5f;
      float x0 = floorf(x), y0 = floorf(y);
      float samp = 0.0f;
      #pragma unroll
      for (int dx = 0; dx < 2; ++dx) {
        #pragma unroll
        for (int dy = 0; dy < 2; ++dy) {
          float xi = x0 + (float)dx, yi = y0 + (float)dy;
          float wgt = (1.0f - fabsf(x - xi)) * (1.0f - fabsf(y - yi));
          if (xi >= 0.0f && xi <= (float)(HW - 1) &&
              yi >= 0.0f && yi <= (float)(HW - 1) && wgt > 0.0f) {
            int vrow = b * QLEN + S + (int)yi * HW + (int)xi;
            float v = (float)value[(size_t)vrow * 256 + h * 32 + lane];
            samp += wgt * v;
          }
        }
      }
      acc += (w16[l * 4 + p] * sinv) * samp;
    }
  }
  attn_out[(size_t)row * 256 + h * 32 + lane] = (bf16_t)acc;
}

// ---------------- LayerNorm: one wave per row of 256 ----------------------
// NOTE: `in` may alias `outF` (in-place LN) -> no restrict on those two.
__global__ __launch_bounds__(256) void layernorm_kernel(
    const float* in, const float* __restrict__ g,
    const float* __restrict__ be, float* outF,
    bf16_t* __restrict__ outB, int rows)
{
  int lane = threadIdx.x & 31;
  int row = blockIdx.x * 8 + (threadIdx.x >> 5);
  if (row >= rows) return;
  const float* x = in + (size_t)row * DMODEL;
  float v[8], s = 0.0f, ss = 0.0f;
  #pragma unroll
  for (int k = 0; k < 8; ++k) {
    float t = x[k * 32 + lane];
    v[k] = t; s += t; ss += t * t;
  }
  #pragma unroll
  for (int o = 16; o > 0; o >>= 1) {
    s  += __shfl_xor(s, o, 32);
    ss += __shfl_xor(ss, o, 32);
  }
  float mean = s * (1.0f / DMODEL);
  float var  = ss * (1.0f / DMODEL) - mean * mean;
  float inv  = rsqrtf(var + EPSLN);
  #pragma unroll
  for (int k = 0; k < 8; ++k) {
    int c = k * 32 + lane;
    float o = (v[k] - mean) * inv * g[c] + be[c];
    outF[(size_t)row * DMODEL + c] = o;
    if (outB) outB[(size_t)row * DMODEL + c] = (bf16_t)o;
  }
}

// ---------------- launch ---------------------------------------------------
extern "C" void kernel_launch(void* const* d_in, const int* in_sizes, int n_in,
                              void* d_out, int out_size, void* d_ws, size_t ws_size,
                              hipStream_t stream) {
  (void)in_sizes; (void)n_in; (void)out_size; (void)ws_size;
  const float* src  = (const float*)d_in[0];
  const float* pos  = (const float*)d_in[1];
  // d_in[2], d_in[3]: static spatial metadata, hardcoded
  const float* Wv   = (const float*)d_in[4];
  const float* bv   = (const float*)d_in[5];
  const float* Woff = (const float*)d_in[6];
  const float* boff = (const float*)d_in[7];
  const float* Wa   = (const float*)d_in[8];
  const float* ba   = (const float*)d_in[9];
  const float* Wo   = (const float*)d_in[10];
  const float* bo   = (const float*)d_in[11];
  const float* W1   = (const float*)d_in[12];
  const float* b1   = (const float*)d_in[13];
  const float* W2   = (const float*)d_in[14];
  const float* b2   = (const float*)d_in[15];
  const float* g1   = (const float*)d_in[16];
  const float* be1  = (const float*)d_in[17];
  const float* g2   = (const float*)d_in[18];
  const float* be2  = (const float*)d_in[19];

  // workspace carve (256B aligned)
  char* base = (char*)d_ws;
  size_t woff = 0;
  auto alloc = [&](size_t bytes) -> void* {
    void* p = base + woff;
    woff = (woff + bytes + 255) & ~(size_t)255;
    return p;
  };
  bf16_t* src_b  = (bf16_t*)alloc((size_t)MROWS * DMODEL * 2);
  bf16_t* q_b    = (bf16_t*)alloc((size_t)MROWS * DMODEL * 2);
  bf16_t* val_b  = (bf16_t*)alloc((size_t)MROWS * DMODEL * 2);
  float*  off_f  = (float*) alloc((size_t)MROWS * DMODEL * 4);
  float*  logit_f= (float*) alloc((size_t)MROWS * 128 * 4);
  bf16_t* attn_b = (bf16_t*)alloc((size_t)MROWS * DMODEL * 2);
  float*  y1     = (float*) alloc((size_t)MROWS * DMODEL * 4); // src+attn, then x after LN1 (in-place)
  bf16_t* x_b    = (bf16_t*)alloc((size_t)MROWS * DMODEL * 2);
  bf16_t* h_b    = (bf16_t*)alloc((size_t)MROWS * DFFN * 2);
  float*  y2     = (float*) alloc((size_t)MROWS * DMODEL * 4);
  bf16_t* WvT    = (bf16_t*)alloc((size_t)256 * 256 * 2);
  bf16_t* WoffT  = (bf16_t*)alloc((size_t)256 * 256 * 2);
  bf16_t* WaT    = (bf16_t*)alloc((size_t)128 * 256 * 2);
  bf16_t* WoT    = (bf16_t*)alloc((size_t)256 * 256 * 2);
  bf16_t* W1T    = (bf16_t*)alloc((size_t)1024 * 256 * 2);
  bf16_t* W2T    = (bf16_t*)alloc((size_t)256 * 1024 * 2);

  const int nElem = MROWS * DMODEL;            // 5,570,560
  prep_kernel<<<nElem / 256, 256, 0, stream>>>(src, pos, src_b, q_b, nElem);

  wtrans_kernel<<<(256*256 + 255)/256, 256, 0, stream>>>(Wv,   WvT,   256, 256);
  wtrans_kernel<<<(256*256 + 255)/256, 256, 0, stream>>>(Woff, WoffT, 256, 256);
  wtrans_kernel<<<(256*128 + 255)/256, 256, 0, stream>>>(Wa,   WaT,   256, 128);
  wtrans_kernel<<<(256*256 + 255)/256, 256, 0, stream>>>(Wo,   WoT,   256, 256);
  wtrans_kernel<<<(256*1024+ 255)/256, 256, 0, stream>>>(W1,   W1T,   256, 1024);
  wtrans_kernel<<<(1024*256+ 255)/256, 256, 0, stream>>>(W2,   W2T,   1024, 256);

  const int gM = MROWS / 64;                   // 340 (block covers 64 rows x 128 cols)
  // value = src @ Wv + bv  (bf16 out for sampling)
  gemm_wmma_kernel<<<dim3(2, gM), 256, 0, stream>>>(
      src_b, WvT, bv, MROWS, 256, 256, nullptr, val_b, nullptr, 0);
  // off = query @ Woff + boff (f32)
  gemm_wmma_kernel<<<dim3(2, gM), 256, 0, stream>>>(
      q_b, WoffT, boff, MROWS, 256, 256, off_f, nullptr, nullptr, 0);
  // attn logits = query @ Wa + ba (f32, N=128)
  gemm_wmma_kernel<<<dim3(1, gM), 256, 0, stream>>>(
      q_b, WaT, ba, MROWS, 128, 256, logit_f, nullptr, nullptr, 0);

  // deformable bilinear sampling + attention-weighted sum -> attn_b
  msdeform_sample_kernel<<<MROWS, 256, 0, stream>>>(val_b, off_f, logit_f, attn_b);

  // y1 = src + attn_b @ Wo + bo
  gemm_wmma_kernel<<<dim3(2, gM), 256, 0, stream>>>(
      attn_b, WoT, bo, MROWS, 256, 256, y1, nullptr, src, 0);
  // x = LN1(y1)  (f32 in-place into y1, bf16 copy for FFN GEMM)
  layernorm_kernel<<<(MROWS + 7) / 8, 256, 0, stream>>>(y1, g1, be1, y1, x_b, MROWS);

  // h = relu(x @ W1 + b1)   (N=1024)
  gemm_wmma_kernel<<<dim3(8, gM), 256, 0, stream>>>(
      x_b, W1T, b1, MROWS, 1024, 256, nullptr, h_b, nullptr, 1);
  // y2 = x + h @ W2 + b2    (K=1024)
  gemm_wmma_kernel<<<dim3(2, gM), 256, 0, stream>>>(
      h_b, W2T, b2, MROWS, 256, 1024, y2, nullptr, y1, 0);
  // out = LN2(y2)
  layernorm_kernel<<<(MROWS + 7) / 8, 256, 0, stream>>>(y2, g2, be2, (float*)d_out, nullptr, MROWS);
}
